// StackedLSTM_36636071035708
// MI455X (gfx1250) — compile-verified, matrix operand order checked by reading, not run on hardware
//
#include <hip/hip_runtime.h>

// Problem dims
#define B_   64
#define T_   256
#define IN_  512
#define H_   1024
#define C_   128
#define G4_  4096   // 4*H

#define LDS_ROW_B 80            // 64B of data (32 bf16) + 16B pad: conflict-free
#define CHUNK_BYTES (64 * LDS_ROW_B)

typedef __bf16 bf16;
typedef unsigned int u32;
typedef __attribute__((ext_vector_type(16))) __bf16 v16bf;
typedef __attribute__((ext_vector_type(8)))  float   v8f;
typedef __attribute__((ext_vector_type(4)))  u32     u32x4;
typedef __attribute__((ext_vector_type(4)))  int     i32x4;

// Pointer types matching the async-LDS builtin prototype:
//   void __builtin_amdgcn_global_load_async_to_lds_b128(i32x4 AS1*, i32x4 AS3*, Ii, Ii)
typedef __attribute__((address_space(1))) i32x4 as1_i32x4;
typedef __attribute__((address_space(3))) i32x4 as3_i32x4;

#define HAS_ASYNC_LDS __has_builtin(__builtin_amdgcn_global_load_async_to_lds_b128)
#define HAS_WAIT_ASYNC __has_builtin(__builtin_amdgcn_s_wait_asynccnt)

union Frag {
  v16bf v;
  u32x4 q[2];
};

__device__ __forceinline__ bf16 f32_to_bf16(float f) {
  union { float f; u32 u; } cv; cv.f = f;
  u32 r = cv.u + 0x7FFFu + ((cv.u >> 16) & 1u);  // round-to-nearest-even
  union { unsigned short s; bf16 b; } o;
  o.s = (unsigned short)(r >> 16);
  return o.b;
}

// ---- B fragment straight from global (weights are L2-resident) ----
// ISA 16-bit 16x32 layout: lane L holds row N=L&15, K in [half*8,+8) and
// [16+half*8,+8), half=L>>4  => two contiguous 16-byte loads.
__device__ __forceinline__ void load_frag_g(Frag& f, const bf16* __restrict__ base,
                                            int ld, int lane) {
  const int m = lane & 15;
  const int half = lane >> 4;
  const bf16* p = base + (size_t)m * (size_t)ld + half * 8;
  f.q[0] = *(const u32x4*)(p);
  f.q[1] = *(const u32x4*)(p + 16);
}

// ---- A fragment from the staged LDS chunk (64 rows x 32 cols bf16) ----
__device__ __forceinline__ void load_frag_lds(Frag& f, const char* buf, int mt, int lane) {
  const int m = (lane & 15) + mt * 16;
  const int half = lane >> 4;
  const char* p = buf + m * LDS_ROW_B + half * 16;
  f.q[0] = *(const u32x4*)(p);        // cols [half*8, half*8+8)
  f.q[1] = *(const u32x4*)(p + 32);   // cols [16+half*8, ...)
}

// ---- stage one 64x32 bf16 A-chunk into LDS (all 128 threads) ----
// 256 x 16B segments; thread tid moves segments tid and tid+128.
__device__ __forceinline__ void stage_chunk(char* dst, const bf16* __restrict__ src,
                                            int lda, int tid) {
#pragma unroll
  for (int i = 0; i < 2; ++i) {
    const int e = tid + i * 128;
    const int row = e >> 2;
    const int q = e & 3;
    const bf16* g = src + (size_t)row * (size_t)lda + q * 8;
    char* l = dst + row * LDS_ROW_B + q * 16;
#if HAS_ASYNC_LDS
    __builtin_amdgcn_global_load_async_to_lds_b128(
        (as1_i32x4*)g, (as3_i32x4*)l, 0, 0);
#else
    *(u32x4*)l = *(const u32x4*)g;    // ds_store_b128 fallback
#endif
  }
}

__device__ __forceinline__ void wait_async_keep2() {
#if HAS_ASYNC_LDS
#if HAS_WAIT_ASYNC
  __builtin_amdgcn_s_wait_asynccnt(2);
#else
  asm volatile("s_wait_asynccnt 0x2" ::: "memory");
#endif
#endif
}
__device__ __forceinline__ void wait_async_all() {
#if HAS_ASYNC_LDS
#if HAS_WAIT_ASYNC
  __builtin_amdgcn_s_wait_asynccnt(0);
#else
  asm volatile("s_wait_asynccnt 0x0" ::: "memory");
#endif
#endif
}

// Accumulate 64(M) x 16(N) strip over K with LDS-staged, double-buffered A.
// acc += A[0:64, 0:K] * W[n0:n0+16, 0:K]^T
__device__ __forceinline__ void mma_panel(v8f acc[4],
                                          const bf16* __restrict__ A, int lda,
                                          const bf16* __restrict__ W, int K,
                                          int n0, int lane, int tid,
                                          char* lds0, char* lds1) {
  const int nchunks = K >> 5;
  stage_chunk(lds0, A, lda, tid);                       // chunk 0
  for (int ck = 0; ck < nchunks; ++ck) {
    char* cur = (ck & 1) ? lds1 : lds0;
    char* nxt = (ck & 1) ? lds0 : lds1;
    const bool more = (ck + 1) < nchunks;
    if (more) {
      stage_chunk(nxt, A + (size_t)(ck + 1) * 32, lda, tid);
      wait_async_keep2();                               // chunk ck landed
    } else {
      wait_async_all();
    }
    __syncthreads();                                    // ck visible to all waves

    Frag bfr;
    load_frag_g(bfr, W + (size_t)n0 * (size_t)K + ck * 32, K, lane);
#pragma unroll
    for (int mt = 0; mt < 4; ++mt) {
      Frag afr;
      load_frag_lds(afr, cur, mt, lane);
      acc[mt] = __builtin_amdgcn_wmma_f32_16x16x32_bf16(
          /*neg_a=*/false, afr.v, /*neg_b=*/false, bfr.v,
          /*c_mod=*/(short)0, acc[mt], /*reuse_a=*/false, /*reuse_b=*/false);
    }
    __syncthreads();                                    // done reading cur
  }
}

// gates[64,4096] = A1[64,K1]*W1^T + A2[64,K2]*W2^T + bias_a + bias_b
// grid.x = 64 blocks, 128 threads (4 waves); wave w owns N strip w*16.
__global__ __launch_bounds__(128) void slstm_gates_gemm(
    const bf16* __restrict__ A1, int lda1, int K1, const bf16* __restrict__ W1,
    const bf16* __restrict__ A2, int K2, const bf16* __restrict__ W2,
    const float* __restrict__ bias_a, const float* __restrict__ bias_b,
    float* __restrict__ gates) {
  __shared__ __align__(16) char smem[2][CHUNK_BYTES];
  const int tid = threadIdx.x;
  const int lane = tid & 31;
  const int wave = tid >> 5;
  const int n0 = blockIdx.x * 64 + wave * 16;

  v8f zv = {0.f, 0.f, 0.f, 0.f, 0.f, 0.f, 0.f, 0.f};
  v8f acc[4];
#pragma unroll
  for (int i = 0; i < 4; ++i) acc[i] = zv;

  mma_panel(acc, A1, lda1, W1, K1, n0, lane, tid, smem[0], smem[1]);
  mma_panel(acc, A2, K2, W2, K2, n0, lane, tid, smem[0], smem[1]);

  const int n = n0 + (lane & 15);
  const int half = lane >> 4;
  const float bias = bias_a[n] + bias_b[n];
#pragma unroll
  for (int mt = 0; mt < 4; ++mt) {
#pragma unroll
    for (int r = 0; r < 8; ++r) {
      // C layout: VGPR r -> row M = r + 8*half, col N = lane&15
      gates[(size_t)(mt * 16 + half * 8 + r) * G4_ + n] = acc[mt][r] + bias;
    }
  }
}

// Elementwise LSTM cell: gate order i,f,g,o along 4H.
__global__ void slstm_cell(const float* __restrict__ gates,
                           float* __restrict__ c,
                           bf16* __restrict__ hb,
                           bf16* __restrict__ Hout) {
  int idx = blockIdx.x * blockDim.x + threadIdx.x;
  if (idx >= B_ * H_) return;
  int b = idx >> 10;            // H_ = 1024
  int n = idx & (H_ - 1);
  const float* g = gates + (size_t)b * G4_;
  float gi = g[n];
  float gf = g[n + H_];
  float gg = g[n + 2 * H_];
  float go = g[n + 3 * H_];
  float si = 1.f / (1.f + __expf(-gi));
  float sf = 1.f / (1.f + __expf(-gf));
  float so = 1.f / (1.f + __expf(-go));
  float ct = sf * c[idx] + si * tanhf(gg);
  float ht = so * tanhf(ct);
  c[idx] = ct;
  bf16 hv = f32_to_bf16(ht);
  hb[idx] = hv;    // recurrent state for next step
  Hout[idx] = hv;  // history for layer-1 input / classifier
}

// logits: Hb[T*B, H] * Wc[C,H]^T + bias -> out[b, n, t*2 + layer]
// grid = (C/64, T), 128 threads; block row-tile = 64 rows = one full t.
__global__ __launch_bounds__(128) void slstm_cls_gemm(
    const bf16* __restrict__ Hb, const bf16* __restrict__ Wc,
    const float* __restrict__ bias, float* __restrict__ out, int layer) {
  __shared__ __align__(16) char smem[2][CHUNK_BYTES];
  const int tid = threadIdx.x;
  const int lane = tid & 31;
  const int wave = tid >> 5;
  const int n0 = blockIdx.x * 64 + wave * 16;
  const int t = blockIdx.y;
  const bf16* A = Hb + (size_t)t * (B_ * H_);

  v8f zv = {0.f, 0.f, 0.f, 0.f, 0.f, 0.f, 0.f, 0.f};
  v8f acc[4];
#pragma unroll
  for (int i = 0; i < 4; ++i) acc[i] = zv;

  mma_panel(acc, A, H_, Wc, H_, n0, lane, tid, smem[0], smem[1]);

  const int n = n0 + (lane & 15);
  const int half = lane >> 4;
  const float bv = bias[n];
#pragma unroll
  for (int mt = 0; mt < 4; ++mt) {
#pragma unroll
    for (int r = 0; r < 8; ++r) {
      int brow = mt * 16 + half * 8 + r;  // batch index
      out[(size_t)brow * (C_ * T_ * 2) + (size_t)n * (T_ * 2) + t * 2 + layer] =
          acc[mt][r] + bv;
    }
  }
}

__global__ void slstm_cvt_bf16(const float* __restrict__ in, bf16* __restrict__ out, int n) {
  int i = blockIdx.x * blockDim.x + threadIdx.x;
  if (i < n) out[i] = f32_to_bf16(in[i]);
}

__global__ void slstm_zero_u32(u32* __restrict__ p, int n) {
  int i = blockIdx.x * blockDim.x + threadIdx.x;
  if (i < n) p[i] = 0u;
}

extern "C" void kernel_launch(void* const* d_in, const int* in_sizes, int n_in,
                              void* d_out, int out_size, void* d_ws, size_t ws_size,
                              hipStream_t stream) {
  (void)in_sizes; (void)n_in; (void)out_size; (void)ws_size;

  const float* xs    = (const float*)d_in[0];
  const float* W_ih0 = (const float*)d_in[1];
  const float* W_hh0 = (const float*)d_in[2];
  const float* b_ih0 = (const float*)d_in[3];
  const float* b_hh0 = (const float*)d_in[4];
  const float* W_ih1 = (const float*)d_in[5];
  const float* W_hh1 = (const float*)d_in[6];
  const float* b_ih1 = (const float*)d_in[7];
  const float* b_hh1 = (const float*)d_in[8];
  const float* W_cls = (const float*)d_in[9];
  const float* b_cls = (const float*)d_in[10];
  float* out = (float*)d_out;

  // ---- workspace carve-out (~112 MB) ----
  char* w = (char*)d_ws;
  auto take = [&](size_t bytes) {
    char* p = w;
    w += (bytes + 255) & ~(size_t)255;
    return p;
  };
  bf16* Xb    = (bf16*)take((size_t)B_ * T_ * IN_ * 2);   // 16 MB
  bf16* Wih0b = (bf16*)take((size_t)G4_ * IN_ * 2);       //  4 MB
  bf16* Whh0b = (bf16*)take((size_t)G4_ * H_ * 2);        //  8 MB
  bf16* Wih1b = (bf16*)take((size_t)G4_ * H_ * 2);        //  8 MB
  bf16* Whh1b = (bf16*)take((size_t)G4_ * H_ * 2);        //  8 MB
  bf16* Wclsb = (bf16*)take((size_t)C_ * H_ * 2);         //  0.25 MB
  bf16* H0b   = (bf16*)take((size_t)T_ * B_ * H_ * 2);    // 32 MB [T,B,H]
  bf16* H1b   = (bf16*)take((size_t)T_ * B_ * H_ * 2);    // 32 MB
  float* c0   = (float*)take((size_t)B_ * H_ * 4);
  float* c1   = (float*)take((size_t)B_ * H_ * 4);
  bf16* h0b   = (bf16*)take((size_t)B_ * H_ * 2);
  bf16* h1b   = (bf16*)take((size_t)B_ * H_ * 2);
  float* gates = (float*)take((size_t)B_ * G4_ * 4);      //  1 MB

  const int TPB = 256;
  auto cdiv = [](int a, int b) { return (a + b - 1) / b; };

  // ---- one-time bf16 conversions ----
  slstm_cvt_bf16<<<cdiv(B_ * T_ * IN_, TPB), TPB, 0, stream>>>(xs, Xb, B_ * T_ * IN_);
  slstm_cvt_bf16<<<cdiv(G4_ * IN_, TPB), TPB, 0, stream>>>(W_ih0, Wih0b, G4_ * IN_);
  slstm_cvt_bf16<<<cdiv(G4_ * H_, TPB), TPB, 0, stream>>>(W_hh0, Whh0b, G4_ * H_);
  slstm_cvt_bf16<<<cdiv(G4_ * H_, TPB), TPB, 0, stream>>>(W_ih1, Wih1b, G4_ * H_);
  slstm_cvt_bf16<<<cdiv(G4_ * H_, TPB), TPB, 0, stream>>>(W_hh1, Whh1b, G4_ * H_);
  slstm_cvt_bf16<<<cdiv(C_ * H_, TPB), TPB, 0, stream>>>(W_cls, Wclsb, C_ * H_);

  // ---- zero initial states ----
  slstm_zero_u32<<<cdiv(B_ * H_, TPB), TPB, 0, stream>>>((u32*)c0, B_ * H_);
  slstm_zero_u32<<<cdiv(B_ * H_, TPB), TPB, 0, stream>>>((u32*)c1, B_ * H_);
  slstm_zero_u32<<<cdiv(B_ * H_ / 2, TPB), TPB, 0, stream>>>((u32*)h0b, B_ * H_ / 2);
  slstm_zero_u32<<<cdiv(B_ * H_ / 2, TPB), TPB, 0, stream>>>((u32*)h1b, B_ * H_ / 2);

  // ---- sequential recurrence (critical path) ----
  for (int t = 0; t < T_; ++t) {
    // layer 0: gates = x_t*Wih0^T + h0*Whh0^T + b
    slstm_gates_gemm<<<dim3(G4_ / 64), 128, 0, stream>>>(
        Xb + (size_t)t * IN_, T_ * IN_, IN_, Wih0b,
        h0b, H_, Whh0b, b_ih0, b_hh0, gates);
    slstm_cell<<<cdiv(B_ * H_, TPB), TPB, 0, stream>>>(
        gates, c0, h0b, H0b + (size_t)t * B_ * H_);

    // layer 1: gates = h0(t)*Wih1^T + h1*Whh1^T + b
    slstm_gates_gemm<<<dim3(G4_ / 64), 128, 0, stream>>>(
        H0b + (size_t)t * B_ * H_, H_, H_, Wih1b,
        h1b, H_, Whh1b, b_ih1, b_hh1, gates);
    slstm_cell<<<cdiv(B_ * H_, TPB), TPB, 0, stream>>>(
        gates, c1, h1b, H1b + (size_t)t * B_ * H_);
  }

  // ---- classifiers: out[b, c, t*2 + layer] ----
  slstm_cls_gemm<<<dim3(C_ / 64, T_), 128, 0, stream>>>(H0b, Wclsb, b_cls, out, 0);
  slstm_cls_gemm<<<dim3(C_ / 64, T_), 128, 0, stream>>>(H1b, Wclsb, b_cls, out, 1);
}
